// HoloLayer_75754633167190
// MI455X (gfx1250) — matrix operand order repaired and checked by reading, not compiled, not run on hardware
//
#include <hip/hip_runtime.h>
#include <math.h>

// ---------------------------------------------------------------------------
// HoloLayer on MI455X (gfx1250): every FFT is folded into bf16 WMMA GEMMs.
//   kfT-weights: AkT = (Wk^T @ F)^T      (DFT folded into projection weights)
//   kf = x @ AkT^T ...                    (bf16 WMMA GEMMs, B in [N,K] layout)
//   mem = cumsum_s(kf*vf); prod = mem*conj(qf)   (fused parallel-scan kernel)
//   out = x + gain * (prod @ FinvT^T)     (irfft-as-GEMM + residual epilogue)
// D=768, rfft bins = 385 -> 770 real cols, padded to NP=800 (K%32==0).
// ---------------------------------------------------------------------------

typedef __attribute__((ext_vector_type(16))) __bf16 v16bf;
typedef __attribute__((ext_vector_type(8)))  __bf16 v8bf;
typedef __attribute__((ext_vector_type(8)))  float  v8f;

#define D_DIM   768
#define S_DIM   4096
#define B_DIM   4
#define M_TOK   (B_DIM * S_DIM)   // 16384
#define NBINS   385
#define NP      800               // padded 2*NBINS

// ---- CDNA5 async global->LDS via inline asm (portable across toolchains) --
// GLOBAL_LOAD_ASYNC_TO_LDS_B128: VDST = 32-bit LDS byte address (generic LDS
// pointers carry the LDS offset in their low 32 bits per the aperture rules),
// VADDR = 64-bit global address.  Tracked by ASYNCcnt.
#define HAVE_ASYNC_LDS 1
__device__ __forceinline__ void async_cp16(const __bf16* g, __bf16* l) {
    const unsigned int lds = (unsigned int)(unsigned long long)l;
    asm volatile("global_load_async_to_lds_b128 %0, %1, off"
                 :: "v"(lds), "v"(g) : "memory");
}
__device__ __forceinline__ void wait_async0() {
#if __has_builtin(__builtin_amdgcn_s_wait_asynccnt)
    __builtin_amdgcn_s_wait_asynccnt(0);
#else
    asm volatile("s_wait_asynccnt 0x0" ::: "memory");
#endif
}

// ---------------------------- small prep kernels ---------------------------

// FT[c][e] = c even: cos(2*pi*f*e/D) ; c odd: -sin(...)   (f = c/2, padded)
__global__ void holo_build_FT(__bf16* __restrict__ FT) {
    int idx = blockIdx.x * blockDim.x + threadIdx.x;
    if (idx >= NP * D_DIM) return;
    int c = idx / D_DIM, e = idx % D_DIM, f = c >> 1;
    float v = 0.f;
    if (f < NBINS) {
        int m = (f * e) % D_DIM;                      // exact angle reduction
        float th = (float)(2.0 * M_PI * (double)m / (double)D_DIM);
        v = (c & 1) ? -sinf(th) : cosf(th);
    }
    FT[idx] = (__bf16)v;
}

// FinvT[n][c]: irfft weights, transposed ([768 x 800]), w_f/D and sign baked.
__global__ void holo_build_FinvT(__bf16* __restrict__ FiT) {
    int idx = blockIdx.x * blockDim.x + threadIdx.x;
    if (idx >= D_DIM * NP) return;
    int n = idx / NP, c = idx % NP, f = c >> 1;
    float v = 0.f;
    if (f < NBINS) {
        float w = (f == 0 || f == D_DIM / 2) ? 1.f : 2.f;
        int m = (f * n) % D_DIM;
        float th = (float)(2.0 * M_PI * (double)m / (double)D_DIM);
        v = (w / (float)D_DIM) * ((c & 1) ? -sinf(th) : cosf(th));
    }
    FiT[idx] = (__bf16)v;
}

// WT[d][e] = W[e][d] cast to bf16 ; z selects which of the 3 weights.
__global__ void holo_transpose_w(const float* __restrict__ Wk,
                                 const float* __restrict__ Wv,
                                 const float* __restrict__ Wq,
                                 __bf16* __restrict__ WkT,
                                 __bf16* __restrict__ WvT,
                                 __bf16* __restrict__ WqT) {
    int idx = blockIdx.x * blockDim.x + threadIdx.x;
    if (idx >= D_DIM * D_DIM) return;
    int d = idx / D_DIM, e = idx % D_DIM;
    const float* W = (blockIdx.z == 0) ? Wk : (blockIdx.z == 1) ? Wv : Wq;
    __bf16* WT = (blockIdx.z == 0) ? WkT : (blockIdx.z == 1) ? WvT : WqT;
    WT[(size_t)d * D_DIM + e] = (__bf16)W[(size_t)e * D_DIM + d];
}

__global__ void holo_cast_x(const float* __restrict__ x, __bf16* __restrict__ xb) {
    size_t idx = (size_t)blockIdx.x * blockDim.x + threadIdx.x;
    if (idx >= (size_t)M_TOK * D_DIM) return;
    xb[idx] = (__bf16)x[idx];
}

// ------------------------- bf16 WMMA tiled GEMM ----------------------------
// C[M,N] = A[M,K] * Bt[N,K]^T.  Both operands are K-contiguous -> both tiles
// stage as contiguous b128 chunks via async global->LDS DMA, double-buffered
// so the copy of tile k+1 overlaps the WMMA of tile k.
// Block tile 128x128x32, 8 wave32 waves, each 32x64 = 2x4 WMMA tiles.
// mode 0: store bf16 C[M,N]   mode 1: Cf = xres + gain[0]*acc (f32)
// mode 2: store bf16 transposed C[N,M] (vectorized b128 stores)
// Requires: M % 128 == 0, K % 32 == 0.  N tail guarded (rows of Bt zeroed).

#define LDT 40  // LDS row stride (elements): conflict-free b128 fragment loads

__global__ __launch_bounds__(256)
void holo_wmma_gemm(const __bf16* __restrict__ A, const __bf16* __restrict__ Bt,
                    __bf16* __restrict__ Cb, float* __restrict__ Cf,
                    const float* __restrict__ xres, const float* __restrict__ gain,
                    int M, int N, int K, int lda, int ldb, int ldc, int mode) {
    __shared__ __align__(64) __bf16 Atile[2][128][LDT];
    __shared__ __align__(64) __bf16 Btile[2][128][LDT];

    const int tid  = threadIdx.x;
    const int lane = tid & 31;
    const int wave = tid >> 5;
    const int mt0  = (wave & 3) * 32;   // wave row offset in block tile
    const int nt0  = (wave >> 2) * 64;  // wave col offset in block tile
    const int mBase = blockIdx.y * 128;
    const int nBase = blockIdx.x * 128;

    // global->LDS assignments: each thread owns one 32B (16 elem) chunk/tile
    const int ar = tid >> 1, ac = (tid & 1) * 16;   // A: 128 rows x 32 K
    const int bn = tid >> 1, bc = (tid & 1) * 16;   // Bt: 128 rows x 32 K
    const bool bvalid = (nBase + bn) < N;

    const __bf16* Arow = A  + (size_t)(mBase + ar) * lda + ac;
    const __bf16* Brow = Bt + (size_t)(nBase + bn) * ldb + bc;

    const float g = (mode == 1) ? gain[0] : 0.f;

    // Bt rows beyond N never get written: zero them once in both buffers.
    if (!bvalid) {
        v8bf z = {};
        *(v8bf*)&Btile[0][bn][bc] = z;  *(v8bf*)&Btile[0][bn][bc + 8] = z;
        *(v8bf*)&Btile[1][bn][bc] = z;  *(v8bf*)&Btile[1][bn][bc + 8] = z;
    }

    v8f acc[2][4] = {};

#if HAVE_ASYNC_LDS
    auto issue = [&](int buf, int kk) {
        async_cp16(Arow + kk,     &Atile[buf][ar][ac]);
        async_cp16(Arow + kk + 8, &Atile[buf][ar][ac + 8]);
        if (bvalid) {
            async_cp16(Brow + kk,     &Btile[buf][bn][bc]);
            async_cp16(Brow + kk + 8, &Btile[buf][bn][bc + 8]);
        }
    };
    issue(0, 0);
#endif

    int buf = 0;
    for (int kk = 0; kk < K; kk += 32, buf ^= 1) {
#if HAVE_ASYNC_LDS
        wait_async0();                       // this wave's copies into `buf` done
        __syncthreads();                     // everyone's copies visible
        if (kk + 32 < K) issue(buf ^ 1, kk + 32);   // overlap DMA with WMMA
#else
        // register-staged fallback (single live buffer)
        const v8bf* ag = (const v8bf*)(Arow + kk);
        v8bf a0 = ag[0], a1 = ag[1];
        v8bf b0 = {}, b1 = {};
        if (bvalid) {
            const v8bf* bg = (const v8bf*)(Brow + kk);
            b0 = bg[0]; b1 = bg[1];
        }
        if (kk + 32 < K) {                   // software prefetch of next K-tile
            __builtin_prefetch(Arow + kk + 32, 0, 1);
            if (bvalid) __builtin_prefetch(Brow + kk + 32, 0, 1);
        }
        __syncthreads();
        *(v8bf*)&Atile[buf][ar][ac]     = a0;
        *(v8bf*)&Atile[buf][ar][ac + 8] = a1;
        if (bvalid) {
            *(v8bf*)&Btile[buf][bn][bc]     = b0;
            *(v8bf*)&Btile[buf][bn][bc + 8] = b1;
        }
        __syncthreads();
#endif
        // ---- fragment gather per ISA 16-bit A/B VGPR layouts ----
        const int mr = lane & 15;
        const int kA = (lane < 16) ? 0 : 8;   // A: half-lane K base
        const int kB = (lane < 16) ? 0 : 8;   // B: half-lane K base
        v16bf afrag[2], bfrag[4];
#pragma unroll
        for (int i = 0; i < 2; ++i) {
            v8bf lo = *(const v8bf*)&Atile[buf][mt0 + i * 16 + mr][kA];
            v8bf hi = *(const v8bf*)&Atile[buf][mt0 + i * 16 + mr][kA + 16];
            afrag[i] = __builtin_shufflevector(lo, hi, 0,1,2,3,4,5,6,7,
                                                        8,9,10,11,12,13,14,15);
        }
#pragma unroll
        for (int j = 0; j < 4; ++j) {
            v8bf lo = *(const v8bf*)&Btile[buf][nt0 + j * 16 + mr][kB];
            v8bf hi = *(const v8bf*)&Btile[buf][nt0 + j * 16 + mr][kB + 16];
            bfrag[j] = __builtin_shufflevector(lo, hi, 0,1,2,3,4,5,6,7,
                                                        8,9,10,11,12,13,14,15);
        }
#pragma unroll
        for (int i = 0; i < 2; ++i)
#pragma unroll
            for (int j = 0; j < 4; ++j)
                acc[i][j] = __builtin_amdgcn_wmma_f32_16x16x32_bf16(
                    false, afrag[i], false, bfrag[j],
                    (short)0, acc[i][j], false, false);
#if HAVE_ASYNC_LDS
        __syncthreads();                     // done reading `buf` before reuse
#endif
    }

    // ---- epilogue: 16x16 f32 C layout -> lane (n = lane&15, m half lane>>4)
    const int nl = lane & 15, mh = (lane >> 4) * 8;
#pragma unroll
    for (int i = 0; i < 2; ++i)
#pragma unroll
        for (int j = 0; j < 4; ++j) {
            const int n = nBase + nt0 + j * 16 + nl;
            if (n >= N) continue;
            const int m = mBase + mt0 + i * 16 + mh;
            if (mode == 2) {                 // transposed store: 8 m's contiguous
                v8bf o;
#pragma unroll
                for (int r = 0; r < 8; ++r) o[r] = (__bf16)acc[i][j][r];
                *(v8bf*)&Cb[(size_t)n * ldc + m] = o;
            } else {
#pragma unroll
                for (int r = 0; r < 8; ++r) {
                    const float v = acc[i][j][r];
                    const size_t o = (size_t)(m + r) * ldc + n;
                    if (mode == 0) Cb[o] = (__bf16)v;
                    else           Cf[o] = xres[o] + g * v;
                }
            }
        }
}

// ------------- fused complex bind + causal scan over S + unbind ------------
// One workgroup per (freq-bin f, batch b): 256 threads x 16 tokens each.
// mem_s = sum_{t<=s} kf_t*vf_t ; prod_s = mem_s * conj(qf_s).  Padded bins
// (f >= 385) carry exact zeros from the padded GEMMs.

__global__ __launch_bounds__(256)
void holo_scan(const __bf16* __restrict__ kf, const __bf16* __restrict__ vf,
               const __bf16* __restrict__ qf, __bf16* __restrict__ prod) {
    const int f = blockIdx.x;            // 0..NP/2-1
    const int b = blockIdx.y;            // 0..B-1
    const int tid = threadIdx.x;
    const size_t base = ((size_t)b * S_DIM) * NP + 2 * f;

    float pr[16], pi[16], qr[16], qi[16];
    float sRe = 0.f, sIm = 0.f;
#pragma unroll
    for (int j = 0; j < 16; ++j) {
        const size_t o = base + (size_t)(tid * 16 + j) * NP;
        const float kr = (float)kf[o], ki = (float)kf[o + 1];
        const float vr = (float)vf[o], vi = (float)vf[o + 1];
        qr[j] = (float)qf[o]; qi[j] = (float)qf[o + 1];
        sRe += kr * vr - ki * vi;        // complex bind kf*vf
        sIm += kr * vi + ki * vr;
        pr[j] = sRe; pi[j] = sIm;        // register-local inclusive scan
    }

    __shared__ float shRe[256], shIm[256];
    shRe[tid] = sRe; shIm[tid] = sIm;
    __syncthreads();
#pragma unroll
    for (int off = 1; off < 256; off <<= 1) {   // Hillis-Steele inclusive scan
        const float r = (tid >= off) ? shRe[tid - off] : 0.f;
        const float i = (tid >= off) ? shIm[tid - off] : 0.f;
        __syncthreads();
        shRe[tid] += r; shIm[tid] += i;
        __syncthreads();
    }
    const float exRe = shRe[tid] - sRe;         // exclusive prefix of this thread
    const float exIm = shIm[tid] - sIm;

#pragma unroll
    for (int j = 0; j < 16; ++j) {
        const float mr = exRe + pr[j], mi = exIm + pi[j];
        const float outR = mr * qr[j] + mi * qi[j];   // mem * conj(q)
        const float outI = mi * qr[j] - mr * qi[j];
        const size_t o = base + (size_t)(tid * 16 + j) * NP;
        prod[o]     = (__bf16)outR;
        prod[o + 1] = (__bf16)outI;
    }
}

// ------------------------------- launcher ----------------------------------

extern "C" void kernel_launch(void* const* d_in, const int* in_sizes, int n_in,
                              void* d_out, int out_size, void* d_ws, size_t ws_size,
                              hipStream_t stream) {
    const float* x    = (const float*)d_in[0];
    const float* Wk   = (const float*)d_in[1];
    const float* Wv   = (const float*)d_in[2];
    const float* Wq   = (const float*)d_in[3];
    const float* gain = (const float*)d_in[4];
    float* out = (float*)d_out;

    // workspace carve-out (~140 MB total)
    char* w = (char*)d_ws;
    auto carve = [&](size_t bytes) -> char* {
        char* p = w; w += (bytes + 255) & ~(size_t)255; return p;
    };
    __bf16* xbf   = (__bf16*)carve((size_t)M_TOK * D_DIM * 2);
    __bf16* WkT   = (__bf16*)carve((size_t)D_DIM * D_DIM * 2);
    __bf16* WvT   = (__bf16*)carve((size_t)D_DIM * D_DIM * 2);
    __bf16* WqT   = (__bf16*)carve((size_t)D_DIM * D_DIM * 2);
    __bf16* FT    = (__bf16*)carve((size_t)NP * D_DIM * 2);
    __bf16* FinvT = (__bf16*)carve((size_t)D_DIM * NP * 2);
    __bf16* AkT   = (__bf16*)carve((size_t)NP * D_DIM * 2);
    __bf16* AvT   = (__bf16*)carve((size_t)NP * D_DIM * 2);
    __bf16* AqT   = (__bf16*)carve((size_t)NP * D_DIM * 2);
    __bf16* kfb   = (__bf16*)carve((size_t)M_TOK * NP * 2);
    __bf16* vfb   = (__bf16*)carve((size_t)M_TOK * NP * 2);
    __bf16* qfb   = (__bf16*)carve((size_t)M_TOK * NP * 2);
    __bf16* prod  = (__bf16*)carve((size_t)M_TOK * NP * 2);

    // --- prep ---
    holo_build_FT   <<<(NP * D_DIM + 255) / 256, 256, 0, stream>>>(FT);
    holo_build_FinvT<<<(D_DIM * NP + 255) / 256, 256, 0, stream>>>(FinvT);
    holo_transpose_w<<<dim3((D_DIM * D_DIM + 255) / 256, 1, 3), 256, 0, stream>>>(
        Wk, Wv, Wq, WkT, WvT, WqT);
    holo_cast_x<<<((size_t)M_TOK * D_DIM + 255) / 256, 256, 0, stream>>>(x, xbf);

    // --- fold DFT into projection weights: AkT = (Wk^T @ F)^T  [800 x 768] ---
    dim3 gSmall((NP + 127) / 128, D_DIM / 128);
    holo_wmma_gemm<<<gSmall, 256, 0, stream>>>(WkT, FT, AkT, nullptr, nullptr, nullptr,
                                               D_DIM, NP, D_DIM, D_DIM, D_DIM, D_DIM, 2);
    holo_wmma_gemm<<<gSmall, 256, 0, stream>>>(WvT, FT, AvT, nullptr, nullptr, nullptr,
                                               D_DIM, NP, D_DIM, D_DIM, D_DIM, D_DIM, 2);
    holo_wmma_gemm<<<gSmall, 256, 0, stream>>>(WqT, FT, AqT, nullptr, nullptr, nullptr,
                                               D_DIM, NP, D_DIM, D_DIM, D_DIM, D_DIM, 2);

    // --- spectral projections: kf/vf/qf = x @ A*^T  [16384 x 800] ---
    dim3 gBig((NP + 127) / 128, M_TOK / 128);
    holo_wmma_gemm<<<gBig, 256, 0, stream>>>(xbf, AkT, kfb, nullptr, nullptr, nullptr,
                                             M_TOK, NP, D_DIM, D_DIM, D_DIM, NP, 0);
    holo_wmma_gemm<<<gBig, 256, 0, stream>>>(xbf, AvT, vfb, nullptr, nullptr, nullptr,
                                             M_TOK, NP, D_DIM, D_DIM, D_DIM, NP, 0);
    holo_wmma_gemm<<<gBig, 256, 0, stream>>>(xbf, AqT, qfb, nullptr, nullptr, nullptr,
                                             M_TOK, NP, D_DIM, D_DIM, D_DIM, NP, 0);

    // --- bind + causal cumsum over S + unbind ---
    holo_scan<<<dim3(NP / 2, B_DIM), 256, 0, stream>>>(kfb, vfb, qfb, prod);

    // --- irfft-as-GEMM + residual epilogue: out = x + gain * (prod @ FinvT^T) ---
    dim3 gOut(D_DIM / 128, M_TOK / 128);
    holo_wmma_gemm<<<gOut, 256, 0, stream>>>(prod, FinvT, nullptr, out, x, gain,
                                             M_TOK, D_DIM, NP, NP, NP, D_DIM, 1);
}
